// SimpleKAN_53661321396587
// MI455X (gfx1250) — matrix operand order repaired and checked by reading, not compiled, not run on hardware
//
#include <hip/hip_runtime.h>

// ---------------------------------------------------------------------------
// SimpleKAN on MI455X (gfx1250):
//   per layer:  C[b,o] = sum_{i,s} A[b, i*12+s] * W[o, i*12+s]
//     A slot 0      = silu(x[b,i])            (bf16)
//     A slots 1..11 = cubic B-spline bases     (bf16, uniform grid h=0.25)
//     W slot 0      = bw[o,i]                  (bf16)
//     W slots 1..11 = sw[o,i,k]*ss[o,i]        (bf16)
//   GEMM: bf16 WMMA (v_wmma_f32_16x16x32_bf16), fp32 accumulate.
//   Wave tile 64x64 (16 WMMA per 16 ds_b128), block tile 256x128,
//   3-stage global->LDS async ring buffer (ASYNCcnt pipeline).
// ---------------------------------------------------------------------------

typedef __attribute__((ext_vector_type(16))) __bf16 v16bf;
typedef __attribute__((ext_vector_type(8)))  __bf16 v8bf;
typedef __attribute__((ext_vector_type(8)))  float  v8f;

#define LDS_STRIDE 40   // shorts per 32-element K row (32 data + 8 pad) = 80 bytes

__device__ __forceinline__ unsigned short f2bf(float f) {
    unsigned u = __builtin_bit_cast(unsigned, f);
    unsigned r = u + 0x7FFFu + ((u >> 16) & 1u);   // round-to-nearest-even
    return (unsigned short)(r >> 16);
}

// ---------------------------------------------------------------------------
// Weight pack: W[o, i*12+s] bf16, row-major [d_out][12*d_in]
// ---------------------------------------------------------------------------
__global__ __launch_bounds__(256) void kan_pack_kernel(
    const float* __restrict__ bw, const float* __restrict__ sw,
    const float* __restrict__ ss, unsigned long long* __restrict__ Wt)
{
    long oi = (long)blockIdx.x * blockDim.x + threadIdx.x;  // o*d_in + i
    float scale = ss[oi];
    const float* sp = sw + oi * 11;
    unsigned short h[12];
    h[0] = f2bf(bw[oi]);
#pragma unroll
    for (int k = 0; k < 11; ++k) h[1 + k] = f2bf(sp[k] * scale);
    unsigned long long p0 = (unsigned long long)h[0] | ((unsigned long long)h[1] << 16)
                          | ((unsigned long long)h[2] << 32) | ((unsigned long long)h[3] << 48);
    unsigned long long p1 = (unsigned long long)h[4] | ((unsigned long long)h[5] << 16)
                          | ((unsigned long long)h[6] << 32) | ((unsigned long long)h[7] << 48);
    unsigned long long p2 = (unsigned long long)h[8] | ((unsigned long long)h[9] << 16)
                          | ((unsigned long long)h[10] << 32) | ((unsigned long long)h[11] << 48);
    Wt[oi * 3 + 0] = p0;
    Wt[oi * 3 + 1] = p1;
    Wt[oi * 3 + 2] = p2;
}

// ---------------------------------------------------------------------------
// Activation expansion: A[b, i*12+s] bf16 (uniform cubic B-spline, 4 nonzero)
// ---------------------------------------------------------------------------
__global__ __launch_bounds__(256) void kan_expand_kernel(
    const float* __restrict__ X, unsigned long long* __restrict__ A)
{
    long tid = (long)blockIdx.x * blockDim.x + threadIdx.x;   // b*d_in + i
    float x = X[tid];
    float s = x * (1.0f / (1.0f + __expf(-x)));               // silu

    float t = (x + 1.75f) * 4.0f;
    t = fminf(fmaxf(t, -100.0f), 100.0f);
    float fj = floorf(t);
    int   j  = (int)fj;
    float u  = t - fj;
    float um = 1.0f - u;
    float u2 = u * u, u3 = u2 * u, um3 = um * um * um;
    const float k6 = 1.0f / 6.0f;
    float c0 = um3 * k6;                                        // m = j-3
    float c1 = (3.0f * u3 - 6.0f * u2 + 4.0f) * k6;             // m = j-2
    float c2 = (-3.0f * u3 + 3.0f * u2 + 3.0f * u + 1.0f) * k6; // m = j-1
    float c3 = u3 * k6;                                         // m = j

    unsigned short h[12];
    h[0] = f2bf(s);
#pragma unroll
    for (int m = 0; m < 11; ++m) {
        int d = j - m;
        float v = (d == 3) ? c0 : (d == 2) ? c1 : (d == 1) ? c2 : (d == 0) ? c3 : 0.0f;
        h[1 + m] = f2bf(v);
    }
    unsigned long long p0 = (unsigned long long)h[0] | ((unsigned long long)h[1] << 16)
                          | ((unsigned long long)h[2] << 32) | ((unsigned long long)h[3] << 48);
    unsigned long long p1 = (unsigned long long)h[4] | ((unsigned long long)h[5] << 16)
                          | ((unsigned long long)h[6] << 32) | ((unsigned long long)h[7] << 48);
    unsigned long long p2 = (unsigned long long)h[8] | ((unsigned long long)h[9] << 16)
                          | ((unsigned long long)h[10] << 32) | ((unsigned long long)h[11] << 48);
    A[tid * 3 + 0] = p0;
    A[tid * 3 + 1] = p1;
    A[tid * 3 + 2] = p2;
}

// ---------------------------------------------------------------------------
// bf16 GEMM:  C[M][N] (f32) = A[M][K] (bf16) * Wt[N][K]^T (bf16)
// Block tile 256x128, 8 waves (4M x 2N), wave tile 64x64, K-step 32.
// 3-stage ring buffer of async global->LDS copies.
// ---------------------------------------------------------------------------
__global__ __launch_bounds__(256) void kan_gemm_kernel(
    const unsigned short* __restrict__ A,    // [M][Kdim] bf16
    const unsigned short* __restrict__ Wt,   // [N][Kdim] bf16
    float* __restrict__ C,                   // [M][N] f32
    int Kdim, int N)
{
    __shared__ unsigned short As[3][256 * LDS_STRIDE];   // 3 x 20 KB
    __shared__ unsigned short Bs[3][128 * LDS_STRIDE];   // 3 x 10 KB

    const int tid   = threadIdx.x;
    const int lane  = tid & 31;
    const int wave  = tid >> 5;
    const int waveM = wave >> 1;   // 0..3 -> 64-row slice
    const int waveN = wave & 1;    // 0..1 -> 64-col slice

    const long mBase = (long)blockIdx.y * 256;
    const long nBase = (long)blockIdx.x * 128;

    // Stage = A 256x32 (16 KB) + B 128x32 (8 KB); 6 x b128 per thread.
    const int rowA0 = tid >> 1;            // 0..127
    const int rowA1 = rowA0 + 128;         // 128..255
    const int off32 = (tid & 1) * 32;      // byte offset inside the 64B K-slice

    const unsigned char* gA0 = (const unsigned char*)(A  + (mBase + rowA0) * (long)Kdim) + off32;
    const unsigned char* gA1 = (const unsigned char*)(A  + (mBase + rowA1) * (long)Kdim) + off32;
    const unsigned char* gB0 = (const unsigned char*)(Wt + (nBase + rowA0) * (long)Kdim) + off32;

    const unsigned rbA0 = (unsigned)(rowA0 * (LDS_STRIDE * 2)) + (unsigned)off32;
    const unsigned rbA1 = (unsigned)(rowA1 * (LDS_STRIDE * 2)) + (unsigned)off32;
    unsigned lA0[3], lA1[3], lB0[3];
#pragma unroll
    for (int s = 0; s < 3; ++s) {
        lA0[s] = (unsigned)(unsigned long long)(&As[s][0]) + rbA0;
        lA1[s] = (unsigned)(unsigned long long)(&As[s][0]) + rbA1;
        lB0[s] = (unsigned)(unsigned long long)(&Bs[s][0]) + rbA0;
    }

    auto issue = [&](int buf, int stage) {
        long kb = (long)stage * 64;   // 32 bf16 = 64 bytes per K-step
        const unsigned char* pa0 = gA0 + kb;
        const unsigned char* pa1 = gA1 + kb;
        const unsigned char* pb0 = gB0 + kb;
        asm volatile(
            "global_load_async_to_lds_b128 %0, %6, off\n\t"
            "global_load_async_to_lds_b128 %1, %7, off\n\t"
            "global_load_async_to_lds_b128 %2, %8, off\n\t"
            "global_load_async_to_lds_b128 %3, %9, off\n\t"
            "global_load_async_to_lds_b128 %4, %10, off\n\t"
            "global_load_async_to_lds_b128 %5, %11, off"
            :: "v"(lA0[buf]), "v"(lA0[buf] + 16u),
               "v"(lA1[buf]), "v"(lA1[buf] + 16u),
               "v"(lB0[buf]), "v"(lB0[buf] + 16u),
               "v"(pa0), "v"(pa0 + 16),
               "v"(pa1), "v"(pa1 + 16),
               "v"(pb0), "v"(pb0 + 16)
            : "memory");
    };

    v8f zero = {0.f, 0.f, 0.f, 0.f, 0.f, 0.f, 0.f, 0.f};
    v8f acc[4][4];
#pragma unroll
    for (int m = 0; m < 4; ++m)
#pragma unroll
        for (int n = 0; n < 4; ++n) acc[m][n] = zero;

    const int nIter = Kdim >> 5;    // >= 192 always

    issue(0, 0);
    issue(1, 1);

    // per-lane fragment addressing (contiguous K: lane half h holds K 16h..16h+15 for both A and B)
    const int aRow0   = waveM * 64 + (lane & 15);
    const int aColOff = (lane & 16) ? 32 : 0;     // bytes: K 0..15 / 16..31 half
    const int bRow0   = waveN * 64 + (lane & 15);
    const int bColOff = (lane & 16) ? 32 : 0;     // bytes: K 0..15 / 16..31 half

    int rd = 0, wr = 2;
    for (int k = 0; k < nIter; ++k) {
        if (k + 2 < nIter) {
            issue(wr, k + 2);
            asm volatile("s_wait_asynccnt 12" ::: "memory");  // stage k landed
        } else if (k + 1 < nIter) {
            asm volatile("s_wait_asynccnt 6" ::: "memory");
        } else {
            asm volatile("s_wait_asynccnt 0" ::: "memory");
        }
        __syncthreads();   // LDS stage `rd` visible to all waves

        // B fragments: 32x16 bf16; lane<16 -> N=lane, K 0..15; lane>=16 -> K 16..31
        v16bf bfrag[4];
#pragma unroll
        for (int n = 0; n < 4; ++n) {
            const unsigned char* p = (const unsigned char*)&Bs[rd][0]
                                   + (bRow0 + n * 16) * (LDS_STRIDE * 2) + bColOff;
            v8bf lo = *(const v8bf*)p;
            v8bf hi = *(const v8bf*)(p + 16);
            bfrag[n] = __builtin_shufflevector(lo, hi, 0,1,2,3,4,5,6,7,8,9,10,11,12,13,14,15);
        }
        // A fragments: 16x32 bf16; lane<16 -> M=lane, K 0..15;
        //              lane>=16 -> M=lane-16, K 16..31
        v16bf af[4];
#pragma unroll
        for (int m = 0; m < 4; ++m) {
            const unsigned char* p = (const unsigned char*)&As[rd][0]
                                   + (aRow0 + m * 16) * (LDS_STRIDE * 2) + aColOff;
            v8bf lo = *(const v8bf*)p;
            v8bf hi = *(const v8bf*)(p + 16);
            af[m] = __builtin_shufflevector(lo, hi, 0,1,2,3,4,5,6,7,8,9,10,11,12,13,14,15);
        }
        __syncthreads();   // all reads done before buffer `rd` is re-filled

#pragma unroll
        for (int m = 0; m < 4; ++m)
#pragma unroll
            for (int n = 0; n < 4; ++n)
                acc[m][n] = __builtin_amdgcn_wmma_f32_16x16x32_bf16(
                    false, af[m], false, bfrag[n], (short)0, acc[m][n], false, false);

        rd = (rd == 2) ? 0 : rd + 1;
        wr = (wr == 2) ? 0 : wr + 1;
    }

    // C/D layout: VGPR v, lane<16 -> M=v, N=lane; lane>=16 -> M=v+8, N=lane-16
#pragma unroll
    for (int m = 0; m < 4; ++m) {
        long row0 = mBase + waveM * 64 + m * 16 + ((lane >> 4) & 1) * 8;
#pragma unroll
        for (int n = 0; n < 4; ++n) {
            long col = nBase + waveN * 64 + n * 16 + (lane & 15);
            float* cp = C + row0 * (long)N + col;
#pragma unroll
            for (int v = 0; v < 8; ++v)
                cp[(long)v * N] = acc[m][n][v];
        }
    }
}

// ---------------------------------------------------------------------------
extern "C" void kernel_launch(void* const* d_in, const int* in_sizes, int n_in,
                              void* d_out, int out_size, void* d_ws, size_t ws_size,
                              hipStream_t stream)
{
    (void)in_sizes; (void)n_in; (void)out_size; (void)ws_size;
    const int B = 8192;
    const int dims[5] = {512, 1024, 1024, 1024, 256};

    const float* x = (const float*)d_in[0];
    const float *bw[4], *sw[4], *ss[4];
    for (int l = 0; l < 4; ++l) {
        bw[l] = (const float*)d_in[1 + 3 * l];
        sw[l] = (const float*)d_in[2 + 3 * l];
        ss[l] = (const float*)d_in[3 + 3 * l];
    }

    char* ws = (char*)d_ws;
    size_t off = 0;
    auto take = [&](size_t bytes) -> char* {
        char* p = ws + off;
        off = (off + bytes + 255) & ~(size_t)255;
        return p;
    };

    unsigned short* Wt[4];
    int Kdim[4];
    for (int l = 0; l < 4; ++l) {
        Kdim[l] = 12 * dims[l];
        Wt[l] = (unsigned short*)take((size_t)dims[l + 1] * Kdim[l] * 2);
    }
    unsigned short* Aexp = (unsigned short*)take((size_t)B * 12288 * 2);
    float* H1 = (float*)take((size_t)B * 1024 * 4);
    float* H2 = (float*)take((size_t)B * 1024 * 4);

    // Pack weights (bf16, [N][K] so the GEMM needs no transpose loads)
    for (int l = 0; l < 4; ++l) {
        long total = (long)dims[l + 1] * dims[l];
        kan_pack_kernel<<<(unsigned)(total / 256), 256, 0, stream>>>(
            bw[l], sw[l], ss[l], (unsigned long long*)Wt[l]);
    }

    const float* cur = x;
    float* hbuf[2] = {H1, H2};
    for (int l = 0; l < 4; ++l) {
        long total = (long)B * dims[l];
        kan_expand_kernel<<<(unsigned)(total / 256), 256, 0, stream>>>(
            cur, (unsigned long long*)Aexp);
        float* outp = (l == 3) ? (float*)d_out : hbuf[l & 1];
        dim3 grid(dims[l + 1] / 128, B / 256);
        kan_gemm_kernel<<<grid, 256, 0, stream>>>(Aexp, Wt[l], outp, Kdim[l], dims[l + 1]);
        cur = outp;
    }
}
